// MINet_53635551593077
// MI455X (gfx1250) — compile-verified, matrix operand order checked by reading, not run on hardware
//
#include <hip/hip_runtime.h>

// ---------------- problem constants ----------------
#define BB   8
#define QQ   4096
#define BQ   (BB * QQ)          // 32768
#define NROWS (4 * BQ)          // 131072 (4 shifts)
#define CAP  NROWS              // per-expert row-list capacity
#define DM   258
#define DMP  288                // DM padded to multiple of 32
#define HID  256
#define EXP  8
#define OUTC 5

typedef __attribute__((ext_vector_type(16))) _Float16 v16h;
typedef __attribute__((ext_vector_type(8)))  _Float16 v8h;
typedef __attribute__((ext_vector_type(8)))  float    v8f;

// Fragment load per CDNA5 16-bit WMMA layout: lane L holds K = kh..kh+7 and
// 16+kh..16+kh+7 (kh = (L>>4)*8) of row (L&15).  Caller passes p = row_base + k + kh.
__device__ __forceinline__ v16h ldfrag(const _Float16* p) {
  union { v16h v; v8h h[2]; } u;
  u.h[0] = *(const v8h*)(p);
  u.h[1] = *(const v8h*)(p + 16);
  return u.v;
}

__device__ __forceinline__ v8f wmma16(v16h a, v16h b, v8f c) {
  return __builtin_amdgcn_wmma_f32_16x16x32_f16(false, a, false, b, (short)0, c,
                                                false, false);
}

// ---------------- conversion kernels ----------------
__global__ void __launch_bounds__(256) k_zero_counts(int* c) {
  if (threadIdx.x < EXP) c[threadIdx.x] = 0;
}

// feat (B,64,48,48) f32 -> f0 [b][y][x][c] f16
__global__ void __launch_bounds__(256) k_cvt_feat0(const float* __restrict__ feat,
                                                   _Float16* __restrict__ f0) {
  int i = blockIdx.x * 256 + threadIdx.x;
  if (i >= BB * 64 * 48 * 48) return;
  int c = i & 63;
  int rest = i >> 6;
  int x = rest % 48; rest /= 48;
  int y = rest % 48;
  int b = rest / 48;
  f0[i] = (_Float16)feat[(((long)b * 64 + c) * 48 + y) * 48 + x];
}

// uw (256,64,3,3) f32 -> wt[o][tap*64+ci] f16  (K = 576)
__global__ void __launch_bounds__(256) k_cvt_convw(const float* __restrict__ uw,
                                                   _Float16* __restrict__ wt) {
  int i = blockIdx.x * 256 + threadIdx.x;
  if (i >= 256 * 576) return;
  int o = i / 576, k = i - o * 576;
  int tap = k >> 6, ci = k & 63;
  int ky = tap / 3, kx = tap % 3;
  wt[i] = (_Float16)uw[((o * 64 + ci) * 3 + ky) * 3 + kx];
}

// src (E,K,N) f32 -> dst (E,Np,Kp) f16 transposed + zero-padded
__global__ void __launch_bounds__(256) k_cvt_tr(const float* __restrict__ src,
                                                _Float16* __restrict__ dst,
                                                int K, int N, int Kp, int Np) {
  int e = blockIdx.y;
  int i = blockIdx.x * 256 + threadIdx.x;
  if (i >= Np * Kp) return;
  int n = i / Kp, k = i - n * Kp;
  float v = (k < K && n < N) ? src[((long)e * K + k) * N + n] : 0.0f;
  dst[(long)e * Np * Kp + i] = (_Float16)v;
}

// ---------------- conv3x3 + pixel-shuffle (implicit GEMM, WMMA) ----------------
// fin  : [b][y][x][64] f16  (H x W)
// fout : [b][2y+i][2x+j][64] f16  (2H x 2W), cout o -> (c=o>>2, i=(o>>1)&1, j=o&1)
__global__ void __launch_bounds__(256) k_conv_ps2(const _Float16* __restrict__ fin,
                                                  _Float16* __restrict__ fout,
                                                  const _Float16* __restrict__ wt,
                                                  const float* __restrict__ bias,
                                                  int H, int W) {
  __shared__ _Float16 As[16 * 576];
  const int tid = threadIdx.x;
  const int tpi = (H * W) >> 4;               // 16-pixel tiles per image
  const int b    = blockIdx.x / tpi;
  const int lin0 = (blockIdx.x % tpi) << 4;

  // stage A-tile (16 pixels x 9 taps x 64 ch) into LDS, zero-fill borders
  for (int task = tid; task < 16 * 72; task += 256) {
    int p = task / 72;
    int r = task - p * 72;
    int tap = r >> 3;
    int c8  = (r & 7) << 3;
    int lin = lin0 + p;
    int y = lin / W + tap / 3 - 1;
    int x = lin % W + tap % 3 - 1;
    v8h val = {};
    if (y >= 0 && y < H && x >= 0 && x < W)
      val = *(const v8h*)(fin + (((long)b * H + y) * W + x) * 64 + c8);
    *(v8h*)(&As[p * 576 + tap * 64 + c8]) = val;
  }
  __syncthreads();

  const int wave = tid >> 5, lane = tid & 31;
  const int kh = (lane >> 4) << 3, khalf = lane >> 4;
  const int o0 = wave * 32 + (lane & 15);
  const int o1 = o0 + 16;
  const _Float16* ar = &As[(lane & 15) * 576];
  const _Float16* w0 = wt + (long)o0 * 576;
  const _Float16* w1 = wt + (long)o1 * 576;
  v8f acc0 = {}, acc1 = {};
  for (int k = 0; k < 576; k += 32) {
    v16h A  = ldfrag(ar + k + kh);
    v16h B0 = ldfrag(w0 + k + kh);
    v16h B1 = ldfrag(w1 + k + kh);
    acc0 = wmma16(A, B0, acc0);
    acc1 = wmma16(A, B1, acc1);
  }
  const float bi0 = bias[o0], bi1 = bias[o1];
  const int c0 = o0 >> 2, s0i = (o0 >> 1) & 1, s0j = o0 & 1;
  const int c1 = o1 >> 2, s1i = (o1 >> 1) & 1, s1j = o1 & 1;
  const int H2 = H * 2, W2 = W * 2;
  const long base = (long)b * H2 * W2;
#pragma unroll
  for (int r = 0; r < 8; ++r) {
    int m   = r + (khalf << 3);
    int lin = lin0 + m;
    int y = lin / W, x = lin % W;
    fout[(base + (long)(2 * y + s0i) * W2 + (2 * x + s0j)) * 64 + c0] =
        (_Float16)(acc0[r] + bi0);
    fout[(base + (long)(2 * y + s1i) * W2 + (2 * x + s1j)) * 64 + c1] =
        (_Float16)(acc1[r] + bi1);
  }
}

// ---------------- row build: gather + gating + scatter ----------------
__global__ void __launch_bounds__(256) k_build(const float* __restrict__ coord,
                                               const _Float16* __restrict__ f0,
                                               const _Float16* __restrict__ f1,
                                               const _Float16* __restrict__ f2,
                                               const _Float16* __restrict__ f3,
                                               const float* __restrict__ gw,
                                               const float* __restrict__ gb,
                                               _Float16* __restrict__ xb,
                                               float* __restrict__ areas,
                                               float* __restrict__ pred,
                                               int* __restrict__ counts,
                                               int* __restrict__ rowl,
                                               float* __restrict__ gatel) {
  int row = blockIdx.x * 256 + threadIdx.x;
  if (row >= NROWS) return;
  int s  = row >> 15;         // shift index (BQ = 2^15)
  int bq = row & (BQ - 1);
  int b  = bq >> 12;          // Q = 2^12
  float cy = coord[bq * 2 + 0];
  float cx = coord[bq * 2 + 1];
  const float rr = 1.0f / 48.0f;
  float vx = (s & 2) ? 1.0f : -1.0f;   // loop order: vx outer, vy inner
  float vy = (s & 1) ? 1.0f : -1.0f;
  float c0 = fminf(fmaxf(cy + vx * rr + 1e-6f, -1.0f + 1e-6f), 1.0f - 1e-6f);
  float c1 = fminf(fmaxf(cx + vy * rr + 1e-6f, -1.0f + 1e-6f), 1.0f - 1e-6f);

  float logits[8];
#pragma unroll
  for (int e = 0; e < 8; ++e) logits[e] = gb[e];

  _Float16* xr = xb + (long)row * DMP;
  int iy0 = 0, ix0 = 0;
  const _Float16* fl[4] = {f0, f1, f2, f3};
  const int hs[4] = {48, 96, 192, 384};
#pragma unroll
  for (int lvl = 0; lvl < 4; ++lvl) {
    const int h = hs[lvl];
    float gy = 0.5f * ((c0 + 1.0f) * h - 1.0f);
    float gx = 0.5f * ((c1 + 1.0f) * h - 1.0f);
    int iy = (int)fminf(fmaxf(floorf(gy + 0.5f), 0.0f), (float)(h - 1));
    int ix = (int)fminf(fmaxf(floorf(gx + 0.5f), 0.0f), (float)(h - 1));
    if (lvl == 0) { iy0 = iy; ix0 = ix; }
    const _Float16* src = fl[lvl] + (((long)b * h + iy) * h + ix) * 64;
#pragma unroll
    for (int c8 = 0; c8 < 8; ++c8) {
      v8h v = *(const v8h*)(src + c8 * 8);
      *(v8h*)(xr + lvl * 64 + c8 * 8) = v;
#pragma unroll
      for (int j = 0; j < 8; ++j) {
        float vf = (float)v[j];
        const float* g = gw + (lvl * 64 + c8 * 8 + j) * 8;
#pragma unroll
        for (int e = 0; e < 8; ++e) logits[e] += vf * g[e];
      }
    }
  }
  float qcy = -1.0f + (2.0f * iy0 + 1.0f) * rr;
  float qcx = -1.0f + (2.0f * ix0 + 1.0f) * rr;
  float rel0 = (cy - qcy) * 48.0f;
  float rel1 = (cx - qcx) * 48.0f;
  xr[256] = (_Float16)rel0;
  xr[257] = (_Float16)rel1;
#pragma unroll
  for (int d = DM; d < DMP; ++d) xr[d] = (_Float16)0.0f;
#pragma unroll
  for (int e = 0; e < 8; ++e)
    logits[e] += rel0 * gw[256 * 8 + e] + rel1 * gw[257 * 8 + e];

  areas[row] = fabsf(rel0 * rel1) + 1e-9f;
#pragma unroll
  for (int n = 0; n < OUTC; ++n) pred[(long)row * OUTC + n] = 0.0f;

  // top-2 + softmax over the two
  int i0 = 0; float v0 = logits[0];
#pragma unroll
  for (int e = 1; e < 8; ++e) if (logits[e] > v0) { v0 = logits[e]; i0 = e; }
  int i1 = 0; float v1 = -3.0e38f;
#pragma unroll
  for (int e = 0; e < 8; ++e) if (e != i0 && logits[e] > v1) { v1 = logits[e]; i1 = e; }
  float e1w = __expf(v1 - v0);
  float z   = 1.0f + e1w;
  float g0 = 1.0f / z, g1 = e1w / z;

  int slot0 = atomicAdd(&counts[i0], 1);
  rowl[(long)i0 * CAP + slot0]  = row;
  gatel[(long)i0 * CAP + slot0] = g0;
  int slot1 = atomicAdd(&counts[i1], 1);
  rowl[(long)i1 * CAP + slot1]  = row;
  gatel[(long)i1 * CAP + slot1] = g1;
}

// ---------------- grouped per-expert MLP (4 WMMA layers) ----------------
__global__ void __launch_bounds__(256) k_expert(const _Float16* __restrict__ xb,
                                                const int* __restrict__ rowl,
                                                const float* __restrict__ gatel,
                                                const int* __restrict__ counts,
                                                const _Float16* __restrict__ w1t,
                                                const float* __restrict__ b1,
                                                const _Float16* __restrict__ w2t,
                                                const float* __restrict__ b2,
                                                const _Float16* __restrict__ w3t,
                                                const float* __restrict__ b3,
                                                const _Float16* __restrict__ w4t,
                                                const float* __restrict__ b4,
                                                float* __restrict__ pred) {
  const int e    = blockIdx.y;
  const int cnt  = counts[e];
  const int tile = blockIdx.x;
  if (tile * 16 >= cnt) return;

  __shared__ _Float16 xs[16 * DMP];
  __shared__ _Float16 hA[16 * 256];
  __shared__ _Float16 hB[16 * 256];
  __shared__ int   rid[16];
  __shared__ float gwt[16];

  const int tid = threadIdx.x;
  if (tid < 16) {
    int idx = tile * 16 + tid;
    int src = idx < cnt ? idx : (tile * 16);     // pad with a valid row, gate 0
    rid[tid] = rowl[(long)e * CAP + src];
    gwt[tid] = idx < cnt ? gatel[(long)e * CAP + idx] : 0.0f;
  }
  __syncthreads();
  for (int task = tid; task < 16 * (DMP / 8); task += 256) {
    int r = task / (DMP / 8);
    int c = (task - r * (DMP / 8)) * 8;
    *(v8h*)(&xs[r * DMP + c]) = *(const v8h*)(xb + (long)rid[r] * DMP + c);
  }
  __syncthreads();

  const int wave = tid >> 5, lane = tid & 31;
  const int n = lane & 15, kh = (lane >> 4) << 3, khalf = lane >> 4;
  const int o0 = wave * 32 + n, o1 = o0 + 16;

  auto layer = [&](const _Float16* Abase, int strideA, int K,
                   const _Float16* Wb, const float* bias, _Float16* Ob) {
    const _Float16* ar = Abase + (lane & 15) * strideA;
    const _Float16* w0 = Wb + (long)o0 * K;
    const _Float16* w1 = Wb + (long)o1 * K;
    v8f a0 = {}, a1 = {};
    for (int k = 0; k < K; k += 32) {
      v16h A  = ldfrag(ar + k + kh);
      v16h B0 = ldfrag(w0 + k + kh);
      v16h B1 = ldfrag(w1 + k + kh);
      a0 = wmma16(A, B0, a0);
      a1 = wmma16(A, B1, a1);
    }
    float bi0 = bias[o0], bi1 = bias[o1];
#pragma unroll
    for (int r = 0; r < 8; ++r) {
      int m = r + (khalf << 3);
      Ob[m * 256 + o0] = (_Float16)fmaxf(a0[r] + bi0, 0.0f);
      Ob[m * 256 + o1] = (_Float16)fmaxf(a1[r] + bi1, 0.0f);
    }
  };

  layer(xs, DMP, DMP, w1t + (long)e * 256 * DMP, b1 + e * 256, hA);
  __syncthreads();
  layer(hA, 256, 256, w2t + (long)e * 256 * 256, b2 + e * 256, hB);
  __syncthreads();
  layer(hB, 256, 256, w3t + (long)e * 256 * 256, b3 + e * 256, hA);
  __syncthreads();

  if (wave == 0) {                 // layer 4: N padded 5->16, single n-tile
    const _Float16* ar = &hA[(lane & 15) * 256];
    const _Float16* w0 = w4t + (long)e * 16 * 256 + (long)n * 256;
    v8f a0 = {};
    for (int k = 0; k < 256; k += 32) {
      v16h A  = ldfrag(ar + k + kh);
      v16h B0 = ldfrag(w0 + k + kh);
      a0 = wmma16(A, B0, a0);
    }
    if (n < OUTC) {
      float bi = b4[e * OUTC + n];
#pragma unroll
      for (int r = 0; r < 8; ++r) {
        int m = r + (khalf << 3);
        if (tile * 16 + m < cnt)
          atomicAdd(&pred[(long)rid[m] * OUTC + n], gwt[m] * (a0[r] + bi));
      }
    }
  }
}

// ---------------- final area blend + bilinear border ----------------
__global__ void __launch_bounds__(256) k_combine(const float* __restrict__ pred,
                                                 const float* __restrict__ areas,
                                                 const float* __restrict__ coord,
                                                 const float* __restrict__ inp,
                                                 float* __restrict__ out) {
  int bq = blockIdx.x * 256 + threadIdx.x;
  if (bq >= BQ) return;
  float a0 = areas[0 * BQ + bq];
  float a1 = areas[1 * BQ + bq];
  float a2 = areas[2 * BQ + bq];
  float a3 = areas[3 * BQ + bq];
  float tot = a0 + a1 + a2 + a3;
  float w0 = a3 / tot, w1 = a2 / tot, w2 = a1 / tot, w3 = a0 / tot;
  float acc[OUTC];
#pragma unroll
  for (int c = 0; c < OUTC; ++c) {
    acc[c] = pred[((long)0 * BQ + bq) * OUTC + c] * w0 +
             pred[((long)1 * BQ + bq) * OUTC + c] * w1 +
             pred[((long)2 * BQ + bq) * OUTC + c] * w2 +
             pred[((long)3 * BQ + bq) * OUTC + c] * w3;
  }
  int b = bq >> 12;
  float cy = coord[bq * 2 + 0];
  float cx = coord[bq * 2 + 1];
  float gy = 0.5f * ((cy + 1.0f) * 48.0f - 1.0f);
  float gx = 0.5f * ((cx + 1.0f) * 48.0f - 1.0f);
  float y0f = floorf(gy), x0f = floorf(gx);
  float wy = gy - y0f, wx = gx - x0f;
  int y0 = min(max((int)y0f, 0), 47);
  int y1 = min(max((int)y0f + 1, 0), 47);
  int x0 = min(max((int)x0f, 0), 47);
  int x1 = min(max((int)x0f + 1, 0), 47);
#pragma unroll
  for (int c = 0; c < OUTC; ++c) {
    const float* pl = inp + ((long)b * OUTC + c) * 48 * 48;
    float v00 = pl[y0 * 48 + x0], v01 = pl[y0 * 48 + x1];
    float v10 = pl[y1 * 48 + x0], v11 = pl[y1 * 48 + x1];
    float bi = v00 * (1 - wy) * (1 - wx) + v01 * (1 - wy) * wx +
               v10 * wy * (1 - wx) + v11 * wy * wx;
    out[(long)bq * OUTC + c] = acc[c] + bi;
  }
}

// ---------------- host launch ----------------
extern "C" void kernel_launch(void* const* d_in, const int* in_sizes, int n_in,
                              void* d_out, int out_size, void* d_ws, size_t ws_size,
                              hipStream_t stream) {
  (void)in_sizes; (void)n_in; (void)out_size; (void)ws_size;
  const float* inp   = (const float*)d_in[0];
  const float* feat  = (const float*)d_in[1];
  const float* coord = (const float*)d_in[2];
  const float* uw1 = (const float*)d_in[4];
  const float* ub1 = (const float*)d_in[5];
  const float* uw2 = (const float*)d_in[6];
  const float* ub2 = (const float*)d_in[7];
  const float* uw3 = (const float*)d_in[8];
  const float* ub3 = (const float*)d_in[9];
  const float* gw  = (const float*)d_in[10];
  const float* gb  = (const float*)d_in[11];
  const float* ew1 = (const float*)d_in[12];
  const float* eb1 = (const float*)d_in[13];
  const float* ew2 = (const float*)d_in[14];
  const float* eb2 = (const float*)d_in[15];
  const float* ew3 = (const float*)d_in[16];
  const float* eb3 = (const float*)d_in[17];
  const float* ew4 = (const float*)d_in[18];
  const float* eb4 = (const float*)d_in[19];
  float* out = (float*)d_out;

  char* p = (char*)d_ws;
  auto alloc = [&](size_t bytes) -> void* {
    void* r = (void*)p;
    p += (bytes + 255) & ~(size_t)255;
    return r;
  };
  _Float16* f0  = (_Float16*)alloc((size_t)BB * 48 * 48 * 64 * 2);
  _Float16* f1  = (_Float16*)alloc((size_t)BB * 96 * 96 * 64 * 2);
  _Float16* f2  = (_Float16*)alloc((size_t)BB * 192 * 192 * 64 * 2);
  _Float16* f3  = (_Float16*)alloc((size_t)BB * 384 * 384 * 64 * 2);
  _Float16* wt1 = (_Float16*)alloc((size_t)256 * 576 * 2);
  _Float16* wt2 = (_Float16*)alloc((size_t)256 * 576 * 2);
  _Float16* wt3 = (_Float16*)alloc((size_t)256 * 576 * 2);
  _Float16* e1t = (_Float16*)alloc((size_t)EXP * 256 * DMP * 2);
  _Float16* e2t = (_Float16*)alloc((size_t)EXP * 256 * 256 * 2);
  _Float16* e3t = (_Float16*)alloc((size_t)EXP * 256 * 256 * 2);
  _Float16* e4t = (_Float16*)alloc((size_t)EXP * 16 * 256 * 2);
  _Float16* xb  = (_Float16*)alloc((size_t)NROWS * DMP * 2);
  float* pred   = (float*)alloc((size_t)NROWS * OUTC * 4);
  float* areas  = (float*)alloc((size_t)NROWS * 4);
  int*   rowl   = (int*)alloc((size_t)EXP * CAP * 4);
  float* gatel  = (float*)alloc((size_t)EXP * CAP * 4);
  int*   counts = (int*)alloc(64);

  k_zero_counts<<<1, 32, 0, stream>>>(counts);
  k_cvt_feat0<<<(BB * 64 * 48 * 48 + 255) / 256, 256, 0, stream>>>(feat, f0);
  k_cvt_convw<<<(256 * 576 + 255) / 256, 256, 0, stream>>>(uw1, wt1);
  k_cvt_convw<<<(256 * 576 + 255) / 256, 256, 0, stream>>>(uw2, wt2);
  k_cvt_convw<<<(256 * 576 + 255) / 256, 256, 0, stream>>>(uw3, wt3);
  k_cvt_tr<<<dim3((256 * DMP + 255) / 256, EXP), 256, 0, stream>>>(ew1, e1t, DM, HID, DMP, 256);
  k_cvt_tr<<<dim3((256 * 256 + 255) / 256, EXP), 256, 0, stream>>>(ew2, e2t, HID, HID, 256, 256);
  k_cvt_tr<<<dim3((256 * 256 + 255) / 256, EXP), 256, 0, stream>>>(ew3, e3t, HID, HID, 256, 256);
  k_cvt_tr<<<dim3((16 * 256 + 255) / 256, EXP), 256, 0, stream>>>(ew4, e4t, HID, OUTC, 256, 16);

  k_conv_ps2<<<BB * 48 * 48 / 16, 256, 0, stream>>>(f0, f1, wt1, ub1, 48, 48);
  k_conv_ps2<<<BB * 96 * 96 / 16, 256, 0, stream>>>(f1, f2, wt2, ub2, 96, 96);
  k_conv_ps2<<<BB * 192 * 192 / 16, 256, 0, stream>>>(f2, f3, wt3, ub3, 192, 192);

  k_build<<<NROWS / 256, 256, 0, stream>>>(coord, f0, f1, f2, f3, gw, gb, xb,
                                           areas, pred, counts, rowl, gatel);

  k_expert<<<dim3(NROWS / 16, EXP), 256, 0, stream>>>(
      xb, rowl, gatel, counts, e1t, eb1, e2t, eb2, e3t, eb3, e4t, eb4, pred);

  k_combine<<<BQ / 256, 256, 0, stream>>>(pred, areas, coord, inp, out);
}
